// ScoreLpPredicter_29317446762533
// MI455X (gfx1250) — compile-verified, matrix operand order checked by reading, not run on hardware
//
#include <hip/hip_runtime.h>

#define D_   512
#define P_   1024
#define E_   256
#define OUT_ 500

typedef __attribute__((ext_vector_type(16))) __bf16 v16bf;
typedef __attribute__((ext_vector_type(8)))  __bf16 v8bf;
typedef __attribute__((ext_vector_type(8)))  float  v8f;

static __device__ __forceinline__ v16bf cat16(v8bf lo, v8bf hi) {
  return __builtin_shufflevector(lo, hi, 0,1,2,3,4,5,6,7,8,9,10,11,12,13,14,15);
}

// ---------------------------------------------------------------------------
// Prep: dst[n*512 + k] = bf16(src[k*ldsrc + n]) for n<Nsrc else 0.  (512x512 out)
// Produces B^T in bf16 so WMMA B-tiles are contiguous 16B loads per lane.
// ---------------------------------------------------------------------------
__global__ __launch_bounds__(256) void transpose_cvt_kernel(
    const float* __restrict__ src, __bf16* __restrict__ dst, int ldsrc, int Nsrc) {
  int idx = blockIdx.x * 256 + threadIdx.x;   // 512*512 total
  int n = idx >> 9;
  int k = idx & 511;
  float v = (n < Nsrc) ? src[(size_t)k * ldsrc + n] : 0.0f;
  dst[(size_t)n * D_ + k] = (__bf16)v;
}

// ---------------------------------------------------------------------------
// Generic small GEMM: C[M,N] = A[M,K] @ B + bias, with B given transposed in
// bf16 (BT[N,K]).  A is fp32, converted to bf16 on the fly.
// WG = 256 thr = 8 waves, tile 64(M) x 128(N); wave = 16(M) x 64(N).
// ---------------------------------------------------------------------------
__global__ __launch_bounds__(256) void gemm_stage_kernel(
    const float* __restrict__ A, const __bf16* __restrict__ BT,
    const float* __restrict__ bias, float* __restrict__ C,
    int M, int N, int K) {
  const int tid   = threadIdx.x;
  const int lane  = tid & 31;
  const int wid   = tid >> 5;
  const int mwave = wid >> 1;          // 0..3
  const int nwave = wid & 1;           // 0..1
  const int m16   = lane & 15;
  const int hi    = lane >> 4;

  const int row0 = blockIdx.x * 64 + mwave * 16;
  const int col0 = blockIdx.y * 128 + nwave * 64;

  v8f acc[4];
  #pragma unroll
  for (int t = 0; t < 4; ++t)
    #pragma unroll
    for (int r = 0; r < 8; ++r) acc[t][r] = 0.0f;

  for (int k0 = 0; k0 < K; k0 += 32) {
    // A tile 16x32: lane m16 row, K = k0 + 8*hi + j (j<8), +16 for upper half
    const float* ap = A + (size_t)(row0 + m16) * K + k0 + 8 * hi;
    v16bf a;
    #pragma unroll
    for (int j = 0; j < 8; ++j) a[j]     = (__bf16)ap[j];
    #pragma unroll
    for (int j = 0; j < 8; ++j) a[8 + j] = (__bf16)ap[16 + j];

    #pragma unroll
    for (int t = 0; t < 4; ++t) {
      const __bf16* bp = BT + (size_t)(col0 + t * 16 + m16) * K + k0 + 8 * hi;
      v16bf b = cat16(*(const v8bf*)bp, *(const v8bf*)(bp + 16));
      acc[t] = __builtin_amdgcn_wmma_f32_16x16x32_bf16(
          false, a, false, b, (short)0, acc[t], false, false);
    }
  }

  #pragma unroll
  for (int t = 0; t < 4; ++t) {
    const int n = col0 + t * 16 + m16;
    const float bv = bias ? bias[n] : 0.0f;
    #pragma unroll
    for (int r = 0; r < 8; ++r) {
      const int m = row0 + r + 8 * hi;
      C[(size_t)m * N + n] = acc[t][r] + bv;
    }
  }
}

// ---------------------------------------------------------------------------
// Fused pairwise kernel:
//   out[p,e,n] = relu( relu(pp[p,:] + ee[e,:] + l1b) @ l2_w + l2b )[n]
// One WG: one p, 128 e's (BM), 256 out cols (BN).  A-panel (128x512 bf16)
// built once in LDS; B staged in double-buffered 32-K slices via
// GLOBAL_LOAD_ASYNC_TO_LDS_B128 (ASYNCcnt), overlapping the WMMA work.
// LDS: 2KB ppb + 1KB l2b + 130KB A (stride 520) + 2x20KB B (stride 40).
// ---------------------------------------------------------------------------
#define BM  128
#define BN  256
#define SAS 520   // sA row stride in bf16 (pad -> conflict-free ds_load_b128)
#define SBS 40    // sB row stride in bf16

#define OFF_L2B  2048
#define OFF_SA   (2048 + 1024)
#define OFF_SB   (OFF_SA + BM * SAS * 2)
#define SB_BYTES (BN * SBS * 2)
#define FUSED_LDS_BYTES (OFF_SB + 2 * SB_BYTES)

__global__ __launch_bounds__(256) void fused_pair_kernel(
    const float* __restrict__ pp, const float* __restrict__ ee,
    const float* __restrict__ l1b, const __bf16* __restrict__ l2wT,
    const float* __restrict__ l2b, float* __restrict__ out) {
  extern __shared__ char smem_raw[];
  float*  s_ppb = (float*)smem_raw;                 // 512 f32
  float*  s_l2b = (float*)(smem_raw + OFF_L2B);     // 256 f32 (padded bias)
  __bf16* sA    = (__bf16*)(smem_raw + OFF_SA);     // 128 x 520

  const int p   = blockIdx.x;
  const int e0  = blockIdx.y * BM;
  const int nb  = blockIdx.z * BN;
  const int tid = threadIdx.x;

  // Per-thread constants for async B staging: thread t owns n-row (nb + t),
  // copying 64 contiguous bytes per K-slice straight into LDS.
  const unsigned voff = (unsigned)(tid * (D_ * 2));                 // global byte off
  const unsigned ldsa0 =
      (unsigned)(uintptr_t)(void*)(smem_raw + OFF_SB) + (unsigned)(tid * (SBS * 2));
  const unsigned long long gbase0 =
      (unsigned long long)(uintptr_t)(l2wT + (size_t)nb * D_);      // + k0*2 per slice

  // Kick off async staging of the first B slice (k0 = 0) into buffer 0.
  asm volatile(
      "global_load_async_to_lds_b128 %0, %1, %2 offset:0\n"
      "global_load_async_to_lds_b128 %0, %1, %2 offset:16\n"
      "global_load_async_to_lds_b128 %0, %1, %2 offset:32\n"
      "global_load_async_to_lds_b128 %0, %1, %2 offset:48"
      :: "v"(ldsa0), "v"(voff), "s"(gbase0) : "memory");

  // ppb = pp[p,:] + l1_b ; padded output bias slice
  for (int k = tid; k < D_; k += 256) s_ppb[k] = pp[(size_t)p * D_ + k] + l1b[k];
  if (tid < BN) {
    const int n = nb + tid;
    s_l2b[tid] = (n < OUT_) ? l2b[n] : 0.0f;
  }
  __syncthreads();

  // A-panel: sA[row,k] = bf16(relu(ppb[k] + ee[e0+row, k]))
  for (int idx = tid; idx < BM * (D_ / 4); idx += 256) {
    const int row = idx >> 7;             // D_/4 == 128
    const int k4  = (idx & 127) * 4;
    const float4 ev = *(const float4*)(ee + (size_t)(e0 + row) * D_ + k4);
    const float4 pv = *(const float4*)(s_ppb + k4);
    __bf16* d = sA + row * SAS + k4;
    d[0] = (__bf16)fmaxf(ev.x + pv.x, 0.0f);
    d[1] = (__bf16)fmaxf(ev.y + pv.y, 0.0f);
    d[2] = (__bf16)fmaxf(ev.z + pv.z, 0.0f);
    d[3] = (__bf16)fmaxf(ev.w + pv.w, 0.0f);
  }

  const int lane  = tid & 31;
  const int wid   = tid >> 5;
  const int mwave = wid >> 1;          // 0..3 : 32 rows each
  const int nwave = wid & 1;           // 0..1 : 128 cols each
  const int m16   = lane & 15;
  const int hi    = lane >> 4;

  v8f acc[2][8];
  #pragma unroll
  for (int mt = 0; mt < 2; ++mt)
    #pragma unroll
    for (int nt = 0; nt < 8; ++nt)
      #pragma unroll
      for (int r = 0; r < 8; ++r) acc[mt][nt][r] = 0.0f;

  int cur = 0;
  for (int k0 = 0; k0 < D_; k0 += 32) {
    // Our async copies for the current buffer must have landed; then barrier
    // so *all* waves' copies have landed and nobody still reads the buffer
    // we are about to refill.
    asm volatile("s_wait_asynccnt 0x0" ::: "memory");
    __syncthreads();

    // Prefetch next K-slice into the other buffer, overlapping the WMMAs.
    if (k0 + 32 < D_) {
      const unsigned ldsa = ldsa0 + (unsigned)((cur ^ 1) * SB_BYTES);
      const unsigned long long gb = gbase0 + (unsigned long long)((k0 + 32) * 2);
      asm volatile(
          "global_load_async_to_lds_b128 %0, %1, %2 offset:0\n"
          "global_load_async_to_lds_b128 %0, %1, %2 offset:16\n"
          "global_load_async_to_lds_b128 %0, %1, %2 offset:32\n"
          "global_load_async_to_lds_b128 %0, %1, %2 offset:48"
          :: "v"(ldsa), "v"(voff), "s"(gb) : "memory");
    }

    const __bf16* sB = (const __bf16*)(smem_raw + OFF_SB + cur * SB_BYTES);

    v16bf a[2];
    #pragma unroll
    for (int mt = 0; mt < 2; ++mt) {
      const __bf16* ap = sA + (mwave * 32 + mt * 16 + m16) * SAS + k0 + 8 * hi;
      a[mt] = cat16(*(const v8bf*)ap, *(const v8bf*)(ap + 16));
    }
    #pragma unroll
    for (int nt = 0; nt < 8; ++nt) {
      const __bf16* bp = sB + (nwave * 128 + nt * 16 + m16) * SBS + 8 * hi;
      const v16bf b = cat16(*(const v8bf*)bp, *(const v8bf*)(bp + 16));
      acc[0][nt] = __builtin_amdgcn_wmma_f32_16x16x32_bf16(
          false, a[0], false, b, (short)0, acc[0][nt], false, false);
      acc[1][nt] = __builtin_amdgcn_wmma_f32_16x16x32_bf16(
          false, a[1], false, b, (short)0, acc[1][nt], false, false);
    }
    cur ^= 1;
  }

  // Epilogue: out[p, e, n] = relu(acc + l2b[n]), drop padded cols n>=500
  #pragma unroll
  for (int nt = 0; nt < 8; ++nt) {
    const int nloc = nwave * 128 + nt * 16 + m16;
    const int n = nb + nloc;
    const float bv = s_l2b[nloc];
    if (n >= OUT_) continue;
    #pragma unroll
    for (int mt = 0; mt < 2; ++mt) {
      const int ebase = e0 + mwave * 32 + mt * 16 + 8 * hi;
      #pragma unroll
      for (int r = 0; r < 8; ++r) {
        const int e = ebase + r;
        out[((size_t)(p * E_ + e)) * OUT_ + n] = fmaxf(acc[mt][nt][r] + bv, 0.0f);
      }
    }
  }
}

// ---------------------------------------------------------------------------
extern "C" void kernel_launch(void* const* d_in, const int* in_sizes, int n_in,
                              void* d_out, int out_size, void* d_ws, size_t ws_size,
                              hipStream_t stream) {
  const float* post  = (const float*)d_in[0];
  const float* emoji = (const float*)d_in[1];
  const float* k1_w  = (const float*)d_in[2];
  const float* k1_b  = (const float*)d_in[3];
  const float* k2_w  = (const float*)d_in[4];
  const float* k2_b  = (const float*)d_in[5];
  const float* l1_w  = (const float*)d_in[6];
  const float* l1_b  = (const float*)d_in[7];
  const float* l2_w  = (const float*)d_in[8];
  const float* l2_b  = (const float*)d_in[9];
  float* out = (float*)d_out;

  char* ws = (char*)d_ws;
  const size_t BF = (size_t)D_ * D_ * 2;   // 512KB per bf16 512x512 matrix
  __bf16* k1wT   = (__bf16*)(ws + 0 * BF);
  __bf16* k2wT   = (__bf16*)(ws + 1 * BF);
  __bf16* w1pT   = (__bf16*)(ws + 2 * BF);
  __bf16* w1eT   = (__bf16*)(ws + 3 * BF);
  __bf16* l2wT   = (__bf16*)(ws + 4 * BF);
  char*   fbase  = ws + 5 * BF;
  float* pbuf  = (float*)(fbase);                                  // 1024x512
  float* ebuf  = (float*)(fbase + (size_t)P_ * D_ * 4);            //  256x512
  float* ppbuf = (float*)(fbase + (size_t)(P_ + E_) * D_ * 4);     // 1024x512
  float* eebuf = (float*)(fbase + (size_t)(2 * P_ + E_) * D_ * 4); //  256x512

  // 1) transpose+convert all weight matrices to bf16 B^T layout
  dim3 tb(256), tg((D_ * D_) / 256);
  transpose_cvt_kernel<<<tg, tb, 0, stream>>>(k1_w, k1wT, 512, 512);
  transpose_cvt_kernel<<<tg, tb, 0, stream>>>(k2_w, k2wT, 512, 512);
  transpose_cvt_kernel<<<tg, tb, 0, stream>>>(l1_w, w1pT, 512, 512);
  transpose_cvt_kernel<<<tg, tb, 0, stream>>>(l1_w + (size_t)D_ * D_, w1eT, 512, 512);
  transpose_cvt_kernel<<<tg, tb, 0, stream>>>(l2_w, l2wT, 500, 500);

  // 2) p = post@k1_w + k1_b ; e = emoji@k2_w + k2_b
  gemm_stage_kernel<<<dim3(P_ / 64, D_ / 128), 256, 0, stream>>>(
      post, k1wT, k1_b, pbuf, P_, D_, D_);
  gemm_stage_kernel<<<dim3(E_ / 64, D_ / 128), 256, 0, stream>>>(
      emoji, k2wT, k2_b, ebuf, E_, D_, D_);

  // 3) pp = p @ l1_w[:D] ; ee = e @ l1_w[D:]   (l1_b folded into fused kernel)
  gemm_stage_kernel<<<dim3(P_ / 64, D_ / 128), 256, 0, stream>>>(
      pbuf, w1pT, nullptr, ppbuf, P_, D_, D_);
  gemm_stage_kernel<<<dim3(E_ / 64, D_ / 128), 256, 0, stream>>>(
      ebuf, w1eT, nullptr, eebuf, E_, D_, D_);

  // 4) fused pairwise relu-GEMM-relu into d_out
  (void)hipFuncSetAttribute(reinterpret_cast<const void*>(fused_pair_kernel),
                            hipFuncAttributeMaxDynamicSharedMemorySize,
                            FUSED_LDS_BYTES);
  fused_pair_kernel<<<dim3(P_, E_ / BM, 512 / BN), 256, FUSED_LDS_BYTES, stream>>>(
      ppbuf, eebuf, l1_b, l2wT, l2_b, out);

  (void)in_sizes; (void)n_in; (void)out_size; (void)ws_size;
}
